// ModAttn_73426760892987
// MI455X (gfx1250) — matrix operand order repaired and checked by reading, not compiled, not run on hardware
//
#include <hip/hip_runtime.h>
#include <hip/hip_bf16.h>

// ---------- types ----------
typedef __attribute__((ext_vector_type(16))) __bf16 bf16x16;
typedef __attribute__((ext_vector_type(8)))  __bf16 bf16x8;
typedef __attribute__((ext_vector_type(4)))  __bf16 bf16x4;
typedef __attribute__((ext_vector_type(8)))  float  f32x8;

#define DIN   1024
#define DCOND 128
#define NF    8
#define NB    2
#define NSEQ  512
#define NHEAD 16
#define HD    64

__device__ __forceinline__ __bf16 to_bf16(float f) { return (__bf16)f; }

// A fragment: 16x32 bf16, row-major source with leading dim ld.
// lane L (m = L&15, hi = L>>4): elems 0..7 = K[hi*8 .. hi*8+7], 8..15 = K[16+hi*8 ..]
__device__ __forceinline__ bf16x16 load_a_frag(const __bf16* base, int ld, int lane) {
    int m = lane & 15, hi = lane >> 4;
    const __bf16* p = base + (size_t)m * ld + hi * 8;
    bf16x8 lo = *(const bf16x8*)(p);
    bf16x8 hv = *(const bf16x8*)(p + 16);
    return __builtin_shufflevector(lo, hv, 0,1,2,3,4,5,6,7,8,9,10,11,12,13,14,15);
}

// B fragment: 32x16 bf16 (K x N), where B[k][n] = src[n][k], src row-major ld.
// lane L (n = L&15, hi = L>>4): elems 0..7 = K[hi*8 .. hi*8+7], 8..15 = K[16+hi*8 ..]
__device__ __forceinline__ bf16x16 load_b_frag(const __bf16* base, int ld, int lane) {
    int n = lane & 15, hi = lane >> 4;
    const __bf16* p = base + (size_t)n * ld + hi * 8;
    bf16x8 lo = *(const bf16x8*)(p);
    bf16x8 hv = *(const bf16x8*)(p + 16);
    return __builtin_shufflevector(lo, hv, 0,1,2,3,4,5,6,7,8,9,10,11,12,13,14,15);
}

__device__ __forceinline__ f32x8 wmma_bf16(bf16x16 a, bf16x16 b, f32x8 c) {
    return __builtin_amdgcn_wmma_f32_16x16x32_bf16(false, a, false, b, (short)0, c, false, false);
}

// ---------- kernel 1: cond = LayerNorm((w_c @ code).T) ----------
__global__ __launch_bounds__(256) void cond_kernel(const float* __restrict__ wc,
                                                   const float* __restrict__ code,
                                                   const float* __restrict__ gamma,
                                                   const float* __restrict__ beta,
                                                   float* __restrict__ cond) {
    int f = blockIdx.x, tid = threadIdx.x;
    float t[4], s = 0.f, s2 = 0.f;
    for (int j = 0; j < 4; ++j) {
        int d = tid + j * 256;
        const float* row = wc + (size_t)d * DCOND;
        float acc = 0.f;
        for (int k = 0; k < DCOND; ++k) acc += row[k] * code[k * NF + f];
        t[j] = acc; s += acc; s2 += acc * acc;
    }
    __shared__ float rs[256], rq[256];
    rs[tid] = s; rq[tid] = s2; __syncthreads();
    for (int off = 128; off > 0; off >>= 1) {
        if (tid < off) { rs[tid] += rs[tid + off]; rq[tid] += rq[tid + off]; }
        __syncthreads();
    }
    float mu  = rs[0] * (1.f / DIN);
    float var = rq[0] * (1.f / DIN) - mu * mu;
    float rstd = rsqrtf(var + 1e-5f);
    for (int j = 0; j < 4; ++j) {
        int d = tid + j * 256;
        cond[f * DIN + d] = (t[j] - mu) * rstd * gamma[d] + beta[d];
    }
}

// ---------- kernel 2: W (first 2048 rows) -> bf16 ----------
__global__ __launch_bounds__(256) void wconv_kernel(const float* __restrict__ W,
                                                    __bf16* __restrict__ wb) {
    size_t idx = ((size_t)blockIdx.x * 256 + threadIdx.x) * 4;
    if (idx >= (size_t)2 * DIN * DIN) return;
    float4 v = *(const float4*)(W + idx);
    bf16x4 o = { to_bf16(v.x), to_bf16(v.y), to_bf16(v.z), to_bf16(v.w) };
    *(bf16x4*)(wb + idx) = o;
}

// ---------- kernel 3: modx[b][f][n][d] = x[b][n][d] * cond[f][d] -> bf16 ----------
__global__ __launch_bounds__(256) void modx_kernel(const float* __restrict__ x,
                                                   const float* __restrict__ cond,
                                                   __bf16* __restrict__ modx) {
    size_t idx = ((size_t)blockIdx.x * 256 + threadIdx.x) * 4;
    int d = (int)(idx & 1023);
    int n = (int)((idx >> 10) & 511);
    int f = (int)((idx >> 19) & 7);
    int b = (int)(idx >> 22);
    float4 xv = *(const float4*)(x + (((size_t)(b * NSEQ + n)) << 10) + d);
    float4 cv = *(const float4*)(cond + ((size_t)f << 10) + d);
    bf16x4 o = { to_bf16(xv.x * cv.x), to_bf16(xv.y * cv.y),
                 to_bf16(xv.z * cv.z), to_bf16(xv.w * cv.w) };
    *(bf16x4*)(modx + idx) = o;
}

// ---------- kernel 4: per (b,f): qk = modx @ W[0:2048]^T + bias; q *= 0.125 ----------
// grid: (M/128=4, E/128=16, B*NF=16), block 256 = 8 waves (4 x 2), wave tile 32x64
__global__ __launch_bounds__(256) void qkv_kernel(const __bf16* __restrict__ modx,
                                                  const __bf16* __restrict__ wb,
                                                  const float* __restrict__ bias,
                                                  __bf16* __restrict__ qws,
                                                  __bf16* __restrict__ kws) {
    int bf = blockIdx.z, b = bf >> 3, f = bf & 7;
    const __bf16* A = modx + (size_t)bf * NSEQ * DIN;
    int mbase = blockIdx.x * 128, ebase = blockIdx.y * 128;
    int tid = threadIdx.x, lane = tid & 31, w = tid >> 5;
    int arow = mbase + (w >> 1) * 32;   // wave's first row
    int ecol = ebase + (w & 1) * 64;    // wave's first out-column

    f32x8 acc[2][4];
    for (int mt = 0; mt < 2; ++mt)
        for (int nt = 0; nt < 4; ++nt)
            acc[mt][nt] = (f32x8){0.f,0.f,0.f,0.f,0.f,0.f,0.f,0.f};

    for (int k0 = 0; k0 < DIN; k0 += 32) {
        bf16x16 af[2], bfg[4];
        af[0] = load_a_frag(A + (size_t)arow * DIN + k0, DIN, lane);
        af[1] = load_a_frag(A + (size_t)(arow + 16) * DIN + k0, DIN, lane);
        for (int nt = 0; nt < 4; ++nt)
            bfg[nt] = load_b_frag(wb + (size_t)(ecol + nt * 16) * DIN + k0, DIN, lane);
        for (int mt = 0; mt < 2; ++mt)
            for (int nt = 0; nt < 4; ++nt)
                acc[mt][nt] = wmma_bf16(af[mt], bfg[nt], acc[mt][nt]);
    }

    int m = lane & 15, hi = lane >> 4;
    for (int nt = 0; nt < 4; ++nt) {
        int e = ecol + nt * 16 + m;             // 0..2047
        float bv = bias[e];
        bool isK = (e >= DIN);
        int eq = isK ? e - DIN : e;
        int h = eq >> 6, hd = eq & 63;
        float sc = isK ? 1.f : 0.125f;          // fold 1/sqrt(hd) into q (exact in bf16)
        __bf16* dst = (isK ? kws : qws)
                    + ((((size_t)b * NHEAD + h) * NF + f) * NSEQ) * HD + hd;
        for (int mt = 0; mt < 2; ++mt)
            for (int r = 0; r < 8; ++r) {
                int row = arow + mt * 16 + r + 8 * hi;
                dst[(size_t)row * HD] = to_bf16((acc[mt][nt][r] + bv) * sc);
            }
    }
}

// ---------- kernel 5: scores = q @ k^T, row softmax, write attn ----------
// grid: (N/16=32, H=16, B*NF=16), block 128 = 4 waves; wave covers 128 key-cols
__global__ __launch_bounds__(128) void attn_kernel(const __bf16* __restrict__ qws,
                                                   const __bf16* __restrict__ kws,
                                                   float* __restrict__ out) {
    int bf = blockIdx.z, b = bf >> 3, f = bf & 7, h = blockIdx.y;
    size_t head = ((size_t)b * NHEAD + h) * NF + f;
    const __bf16* q = qws + head * NSEQ * HD;
    const __bf16* k = kws + head * NSEQ * HD;
    int mbase = blockIdx.x * 16;
    int tid = threadIdx.x, lane = tid & 31, w = tid >> 5;

    __shared__ float smem[16 * 520];
    __shared__ float red[16][8];
    __shared__ float rinv[16];

    bf16x16 aq[2];
    aq[0] = load_a_frag(q + (size_t)mbase * HD + 0,  HD, lane);
    aq[1] = load_a_frag(q + (size_t)mbase * HD + 32, HD, lane);

    int m = lane & 15, hi = lane >> 4;
    for (int nt = 0; nt < 8; ++nt) {
        int col = w * 128 + nt * 16;
        f32x8 acc = (f32x8){0.f,0.f,0.f,0.f,0.f,0.f,0.f,0.f};
        acc = wmma_bf16(aq[0], load_b_frag(k + (size_t)col * HD + 0,  HD, lane), acc);
        acc = wmma_bf16(aq[1], load_b_frag(k + (size_t)col * HD + 32, HD, lane), acc);
        for (int r = 0; r < 8; ++r)
            smem[(r + 8 * hi) * 520 + col + m] = acc[r];
    }
    __syncthreads();

    int row = tid >> 3, seg = tid & 7;
    float* srow = smem + row * 520 + seg * 64;
    float mx = -3.4e38f;
    for (int c = 0; c < 64; ++c) mx = fmaxf(mx, srow[c]);
    red[row][seg] = mx;
    __syncthreads();
    float rmax = red[row][0];
    for (int j = 1; j < 8; ++j) rmax = fmaxf(rmax, red[row][j]);
    __syncthreads();
    float ls = 0.f;
    for (int c = 0; c < 64; ++c) {
        float e = __expf(srow[c] - rmax);
        srow[c] = e; ls += e;
    }
    red[row][seg] = ls;
    __syncthreads();
    float tot = 0.f;
    for (int j = 0; j < 8; ++j) tot += red[row][j];
    if (seg == 0) rinv[row] = 1.f / tot;
    __syncthreads();

    // coalesced output: each thread writes float4 per row
    for (int r = 0; r < 16; ++r) {
        float inv = rinv[r];
        float4 v = *(const float4*)(smem + r * 520 + tid * 4);
        float4 o; o.x = v.x * inv; o.y = v.y * inv; o.z = v.z * inv; o.w = v.w * inv;
        *(float4*)(out + (head * NSEQ + (size_t)(mbase + r)) * NSEQ + tid * 4) = o;
    }
}

// ---------- launch ----------
extern "C" void kernel_launch(void* const* d_in, const int* in_sizes, int n_in,
                              void* d_out, int out_size, void* d_ws, size_t ws_size,
                              hipStream_t stream) {
    const float* x     = (const float*)d_in[0];
    const float* code  = (const float*)d_in[1];
    const float* wc    = (const float*)d_in[2];
    const float* W     = (const float*)d_in[3];
    const float* bias  = (const float*)d_in[4];
    const float* gamma = (const float*)d_in[5];
    const float* beta  = (const float*)d_in[6];

    char* ws = (char*)d_ws;
    float*  cond = (float*)ws;                          // 8*1024*4      = 32 KB
    __bf16* wb   = (__bf16*)(ws + 32768);               // 2048*1024*2   = 4 MB
    __bf16* modx = (__bf16*)(ws + 4227072);             // 16*512*1024*2 = 16 MB
    __bf16* qws  = (__bf16*)(ws + 21004288);            // 16 MB
    __bf16* kws  = (__bf16*)(ws + 37781504);            // 16 MB  (total ~54.6 MB)

    cond_kernel <<<8,    256, 0, stream>>>(wc, code, gamma, beta, cond);
    wconv_kernel<<<2048, 256, 0, stream>>>(W, wb);
    modx_kernel <<<8192, 256, 0, stream>>>(x, cond, modx);
    qkv_kernel  <<<dim3(4, 16, 16),  256, 0, stream>>>(modx, wb, bias, qws, kws);
    attn_kernel <<<dim3(32, 16, 16), 128, 0, stream>>>(qws, kws, (float*)d_out);
}
